// DMN_32255204393416
// MI455X (gfx1250) — compile-verified, matrix operand order checked by reading, not run on hardware
//
#include <hip/hip_runtime.h>
#include <math.h>

// ---------------------------------------------------------------------------
// CDNA5 (gfx1250) DMN forward. wave32, WMMA bf16 -> f32.
// ---------------------------------------------------------------------------

typedef __attribute__((ext_vector_type(16))) __bf16 v16bf;
typedef __attribute__((ext_vector_type(8)))  float  v8f;

#define HD    256
#define BATCH 64
#define NSENT 40
#define SLEN  30
#define QLEN  20
#define ALEN  12
#define G3    (3*HD)    // 768
#define G7    (7*HD)    // 1792

#define LOG2E 1.4426950408889634f

// Fast transcendentals: v_exp_f32 (2^x) + v_rcp_f32, avoids libm div/branch bloat
__device__ __forceinline__ float sigmoidf_(float x) {
  float e = __builtin_amdgcn_exp2f(-LOG2E * x);
  return __builtin_amdgcn_rcpf(1.0f + e);
}
__device__ __forceinline__ float tanhf_(float x) {
  float xc = fminf(fmaxf(x, -15.0f), 15.0f);
  float e = __builtin_amdgcn_exp2f((2.0f * LOG2E) * xc);
  return (e - 1.0f) * __builtin_amdgcn_rcpf(e + 1.0f);
}

// ISA 16-bit A/B fragment K mapping for 16x32 tiles (per lane-half)
__device__ __forceinline__ int kmap_(int e, int half) {
  int v = e >> 1, p = e & 1;
  int base = (v < 4) ? (v * 2 + half * 8) : (16 + (v - 4) * 2 + half * 8);
  return base + p;
}

__device__ __forceinline__ v8f wmma_bf16(v16bf a, v16bf b, v8f c) {
  return __builtin_amdgcn_wmma_f32_16x16x32_bf16(false, a, false, b, (short)0, c, false, false);
}

// ---------------------------------------------------------------------------
// f32 -> bf16 weight conversion
// ---------------------------------------------------------------------------
__global__ __launch_bounds__(256) void to_bf16_k(const float* __restrict__ src,
                                                 __bf16* __restrict__ dst, int n) {
  int i = blockIdx.x * 256 + threadIdx.x;
  if (i < n) dst[i] = (__bf16)src[i];
}

// ---------------------------------------------------------------------------
// Fused GRU scan. Each workgroup owns 32 sequences; hidden state in LDS.
// Per step: gi = x_t @ Wih^T, gh = h @ Whh^T via WMMA (6 gate tiles/wave),
// then GRU nonlinearity in registers. Captures h at t == len-1 (mask zeros).
// ---------------------------------------------------------------------------
#define SROWS 32
__global__ __launch_bounds__(512)
void gru_scan_k(const int* __restrict__ tok, const int* __restrict__ mask,
                const float* __restrict__ embed,
                const __bf16* __restrict__ Wih, const __bf16* __restrict__ Whh,
                const float* __restrict__ bih, const float* __restrict__ bhh,
                float* __restrict__ enc_out, int rows, int T) {
  __shared__ float  h_s[SROWS][HD];   // 32 KB
  __shared__ __bf16 x_s[SROWS][HD];   // 16 KB
  __shared__ int    cap_s[SROWS];

  const int tid  = threadIdx.x;
  const int lane = tid & 31;
  const int wave = tid >> 5;          // 0..15
  const int half = lane >> 4;
  const int mloc = lane & 15;
  const int row0 = blockIdx.x * SROWS;

  // Pull this wave's recurrent weight rows toward the WGP (global_prefetch_b8)
  {
    const int bn = wave * 16 + mloc;
#pragma unroll
    for (int g = 0; g < 3; ++g) {
      __builtin_prefetch(&Wih[(size_t)(g * HD + bn) * HD], 0, 1);
      __builtin_prefetch(&Whh[(size_t)(g * HD + bn) * HD], 0, 1);
    }
  }

  for (int i = tid; i < SROWS * HD; i += 512) h_s[i / HD][i % HD] = 0.0f;
  if (tid < SROWS) {
    int r = row0 + tid, len = 0;
    if (r < rows)
      for (int t = 0; t < T; ++t) len += (mask[r * T + t] == 0);
    int c = len - 1;
    if (c < 0) c = 0;
    if (c > T - 1) c = T - 1;
    cap_s[tid] = c;
  }
  __syncthreads();

  for (int t = 0; t < T; ++t) {
    // gather x_t = embed[tokens]
    for (int i = tid; i < SROWS * HD; i += 512) {
      int rr = i / HD, k = i % HD;
      int gr = row0 + rr;
      int id = (gr < rows) ? tok[gr * T + t] : 0;
      x_s[rr][k] = (__bf16)embed[(size_t)id * HD + k];
    }
    __syncthreads();

    for (int mt = 0; mt < 2; ++mt) {         // row-tile pass (rows mt*16..)
      const int nt = wave;                   // 16 col tiles cover 256 dims
      const int am = mt * 16 + mloc;
      const int bn = nt * 16 + mloc;
      v8f air = {}, aiz = {}, ain = {}, ahr = {}, ahz = {}, ahn = {};
      for (int kb = 0; kb < HD / 32; ++kb) {
        const int k0 = kb * 32;
        v16bf ax, ah;
#pragma unroll
        for (int e = 0; e < 16; ++e) {
          int k = k0 + kmap_(e, half);
          ax[e] = x_s[am][k];
          ah[e] = (__bf16)h_s[am][k];
        }
        v16bf b0, b1, b2, c0, c1, c2;
#pragma unroll
        for (int e = 0; e < 16; ++e) {
          int k = k0 + kmap_(e, half);
          b0[e] = Wih[(0 * HD + bn) * HD + k];
          b1[e] = Wih[(1 * HD + bn) * HD + k];
          b2[e] = Wih[(2 * HD + bn) * HD + k];
          c0[e] = Whh[(0 * HD + bn) * HD + k];
          c1[e] = Whh[(1 * HD + bn) * HD + k];
          c2[e] = Whh[(2 * HD + bn) * HD + k];
        }
        air = wmma_bf16(ax, b0, air);
        aiz = wmma_bf16(ax, b1, aiz);
        ain = wmma_bf16(ax, b2, ain);
        ahr = wmma_bf16(ah, c0, ahr);
        ahz = wmma_bf16(ah, c1, ahz);
        ahn = wmma_bf16(ah, c2, ahn);
      }
      __syncthreads();   // all waves done reading h rows of this tile
      const int j = nt * 16 + mloc;
      const float bir = bih[j], biz = bih[HD + j], bin = bih[2 * HD + j];
      const float bhr = bhh[j], bhz = bhh[HD + j], bhn = bhh[2 * HD + j];
#pragma unroll
      for (int e = 0; e < 8; ++e) {
        int m = mt * 16 + e + 8 * half;
        float r  = sigmoidf_(air[e] + bir + ahr[e] + bhr);
        float z  = sigmoidf_(aiz[e] + biz + ahz[e] + bhz);
        float nn = tanhf_(ain[e] + bin + r * (ahn[e] + bhn));
        float hold = h_s[m][j];
        float hnew = (1.0f - z) * nn + z * hold;
        h_s[m][j] = hnew;
        int gr = row0 + m;
        if (gr < rows && t == cap_s[m]) enc_out[gr * HD + j] = hnew;
      }
      __syncthreads();
    }
  }
}

// ---------------------------------------------------------------------------
// out[M,N] = A[M,K](f32) @ W^T (W bf16 [N,K]) + bias[N].
// 128-thread blocks (4 waves); each wave computes one 16x16 tile.
// ---------------------------------------------------------------------------
__global__ __launch_bounds__(128)
void gemm_bias_k(const float* __restrict__ A, const __bf16* __restrict__ W,
                 const float* __restrict__ bias, float* __restrict__ out,
                 int M, int N, int K) {
  const int tid  = threadIdx.x;
  const int lane = tid & 31;
  const int wave = tid >> 5;  // 0..3
  const int half = lane >> 4, mloc = lane & 15;
  const int m0 = blockIdx.x * 16;
  const int n0 = (blockIdx.y * 4 + wave) * 16;
  v8f acc = {};
  for (int k0 = 0; k0 < K; k0 += 32) {
    v16bf a, b;
#pragma unroll
    for (int e = 0; e < 16; ++e) {
      int k = k0 + kmap_(e, half);
      a[e] = (__bf16)A[(size_t)(m0 + mloc) * K + k];
      b[e] = W[(size_t)(n0 + mloc) * K + k];
    }
    acc = wmma_bf16(a, b, acc);
  }
#pragma unroll
  for (int e = 0; e < 8; ++e) {
    int m = m0 + e + 8 * half, n = n0 + mloc;
    out[(size_t)m * N + n] = acc[e] + bias[n];
  }
}

// ---------------------------------------------------------------------------
// Episodic memory: single persistent workgroup, 3 episodes x 40 steps.
// Gate features synthesized on the fly from LDS state; all GEMMs via WMMA.
// ---------------------------------------------------------------------------
__device__ __forceinline__ float feat_val(int m, int kk, int s,
                                          const float* __restrict__ enc_ctx,
                                          const float (*mem_s)[HD],
                                          const __bf16 (*q_s)[HD]) {
  int seg = kk >> 8;       // /256
  int j   = kk & 255;
  float ct = enc_ctx[(size_t)(m * NSENT + s) * HD + j];
  float qv = (float)q_s[m][j];
  float mv = mem_s[m][j];
  switch (seg) {
    case 0: return ct;
    case 1: return mv;
    case 2: return qv;
    case 3: return ct * qv;
    case 4: return ct * mv;
    case 5: return fabsf(ct - qv);
    default: return fabsf(ct - mv);
  }
}

__global__ __launch_bounds__(512)
void episodic_k(const float* __restrict__ enc_ctx,  // [B*NS, H]
                const float* __restrict__ enc_q,    // [B, H]
                const float* __restrict__ attn_gi,  // [B*NS, 3H] incl. bih
                const __bf16* __restrict__ gateW1,  // [H, 7H]
                const float* __restrict__ gate_b1, const float* __restrict__ gate_W2,
                const float* __restrict__ gate_b2,
                const __bf16* __restrict__ attn_Whh, const float* __restrict__ attn_bhh,
                const __bf16* __restrict__ mem_Wih, const __bf16* __restrict__ mem_Whh,
                const float* __restrict__ mem_bih, const float* __restrict__ mem_bhh,
                float* __restrict__ mem_out) {
  __shared__ float  mem_s[BATCH][HD];   // 64 KB
  __shared__ float  he_s[BATCH][HD];    // 64 KB
  __shared__ __bf16 q_s[BATCH][HD];     // 32 KB
  __shared__ __bf16 g1_s[BATCH][HD];    // 32 KB
  __shared__ float  g_s[BATCH];

  const int tid  = threadIdx.x;
  const int lane = tid & 31;
  const int wave = tid >> 5;   // 0..15 -> col tile
  const int half = lane >> 4;
  const int mloc = lane & 15;

  for (int i = tid; i < BATCH * HD; i += 512) {
    float v = enc_q[i];
    mem_s[i / HD][i % HD] = v;
    q_s[i / HD][i % HD] = (__bf16)v;
  }
  __syncthreads();

  for (int ep = 0; ep < 3; ++ep) {
    for (int i = tid; i < BATCH * HD; i += 512) he_s[i / HD][i % HD] = 0.0f;
    __syncthreads();

    for (int s = 0; s < NSENT; ++s) {
      // ---- gate layer 1: g1 = tanh(feat @ gate_W1^T + b1), feat K=1792
      for (int mt = 0; mt < 4; ++mt) {
        const int am = mt * 16 + mloc;
        const int bn = wave * 16 + mloc;
        v8f acc = {};
        for (int kb = 0; kb < G7 / 32; ++kb) {
          const int k0 = kb * 32;
          v16bf a, b;
#pragma unroll
          for (int e = 0; e < 16; ++e) {
            int kk = k0 + kmap_(e, half);
            a[e] = (__bf16)feat_val(am, kk, s, enc_ctx, mem_s, q_s);
            b[e] = gateW1[(size_t)bn * G7 + kk];
          }
          acc = wmma_bf16(a, b, acc);
        }
        const int j = wave * 16 + mloc;
#pragma unroll
        for (int e = 0; e < 8; ++e) {
          int m = mt * 16 + e + 8 * half;
          g1_s[m][j] = (__bf16)tanhf_(acc[e] + gate_b1[j]);
        }
      }
      __syncthreads();
      // ---- gate layer 2 + sigmoid (tiny)
      if (tid < BATCH) {
        float a = gate_b2[0];
        for (int k = 0; k < HD; ++k) a += (float)g1_s[tid][k] * gate_W2[k];
        g_s[tid] = sigmoidf_(a);
      }
      __syncthreads();
      // ---- attention GRU cell + gated blend
      for (int mt = 0; mt < 4; ++mt) {
        const int am = mt * 16 + mloc;
        const int bn = wave * 16 + mloc;
        v8f ahr = {}, ahz = {}, ahn = {};
        for (int kb = 0; kb < HD / 32; ++kb) {
          const int k0 = kb * 32;
          v16bf a, c0, c1, c2;
#pragma unroll
          for (int e = 0; e < 16; ++e) {
            int k = k0 + kmap_(e, half);
            a[e]  = (__bf16)he_s[am][k];
            c0[e] = attn_Whh[(0 * HD + bn) * HD + k];
            c1[e] = attn_Whh[(1 * HD + bn) * HD + k];
            c2[e] = attn_Whh[(2 * HD + bn) * HD + k];
          }
          ahr = wmma_bf16(a, c0, ahr);
          ahz = wmma_bf16(a, c1, ahz);
          ahn = wmma_bf16(a, c2, ahn);
        }
        __syncthreads();
        const int j = wave * 16 + mloc;
        const float bhr = attn_bhh[j], bhz = attn_bhh[HD + j], bhn = attn_bhh[2 * HD + j];
#pragma unroll
        for (int e = 0; e < 8; ++e) {
          int m = mt * 16 + e + 8 * half;
          const float* gi = attn_gi + (size_t)(m * NSENT + s) * G3;
          float r  = sigmoidf_(gi[j] + ahr[e] + bhr);
          float z  = sigmoidf_(gi[HD + j] + ahz[e] + bhz);
          float nn = tanhf_(gi[2 * HD + j] + r * (ahn[e] + bhn));
          float hold = he_s[m][j];
          float hc = (1.0f - z) * nn + z * hold;
          float g = g_s[m];
          he_s[m][j] = g * hc + (1.0f - g) * hold;
        }
        __syncthreads();
      }
    }

    // ---- memory GRU step: memory = GRU(memory, e)
    for (int mt = 0; mt < 4; ++mt) {
      const int am = mt * 16 + mloc;
      const int bn = wave * 16 + mloc;
      v8f air = {}, aiz = {}, ain = {}, ahr = {}, ahz = {}, ahn = {};
      for (int kb = 0; kb < HD / 32; ++kb) {
        const int k0 = kb * 32;
        v16bf ae, am2, b0, b1, b2, c0, c1, c2;
#pragma unroll
        for (int e = 0; e < 16; ++e) {
          int k = k0 + kmap_(e, half);
          ae[e]  = (__bf16)he_s[am][k];
          am2[e] = (__bf16)mem_s[am][k];
          b0[e] = mem_Wih[(0 * HD + bn) * HD + k];
          b1[e] = mem_Wih[(1 * HD + bn) * HD + k];
          b2[e] = mem_Wih[(2 * HD + bn) * HD + k];
          c0[e] = mem_Whh[(0 * HD + bn) * HD + k];
          c1[e] = mem_Whh[(1 * HD + bn) * HD + k];
          c2[e] = mem_Whh[(2 * HD + bn) * HD + k];
        }
        air = wmma_bf16(ae, b0, air);
        aiz = wmma_bf16(ae, b1, aiz);
        ain = wmma_bf16(ae, b2, ain);
        ahr = wmma_bf16(am2, c0, ahr);
        ahz = wmma_bf16(am2, c1, ahz);
        ahn = wmma_bf16(am2, c2, ahn);
      }
      __syncthreads();
      const int j = wave * 16 + mloc;
      const float bir = mem_bih[j], biz = mem_bih[HD + j], bin = mem_bih[2 * HD + j];
      const float bhr = mem_bhh[j], bhz = mem_bhh[HD + j], bhn = mem_bhh[2 * HD + j];
#pragma unroll
      for (int e = 0; e < 8; ++e) {
        int m = mt * 16 + e + 8 * half;
        float r  = sigmoidf_(air[e] + bir + ahr[e] + bhr);
        float z  = sigmoidf_(aiz[e] + biz + ahz[e] + bhz);
        float nn = tanhf_(ain[e] + bin + r * (ahn[e] + bhn));
        float hold = mem_s[m][j];
        mem_s[m][j] = (1.0f - z) * nn + z * hold;
      }
      __syncthreads();
    }
  }

  for (int i = tid; i < BATCH * HD; i += 512) mem_out[i] = mem_s[i / HD][i % HD];
}

// ---------------------------------------------------------------------------
// Output head (small, f32 scalar): hidden = tanh([mem,a1,a2] @ W1^T + b1),
// logits = hidden @ W2^T + b2, softmax.
// ---------------------------------------------------------------------------
__global__ __launch_bounds__(512)
void final_head_k(const float* __restrict__ mem, const float* __restrict__ a1,
                  const float* __restrict__ a2,
                  const float* __restrict__ W1, const float* __restrict__ b1,
                  const float* __restrict__ W2, const float* __restrict__ b2,
                  float* __restrict__ out) {
  __shared__ float hid[BATCH][HD];
  const int tid = threadIdx.x;
  for (int idx = tid; idx < BATCH * HD; idx += 512) {
    int m = idx / HD, j = idx % HD;
    const float* w = W1 + (size_t)j * G3;
    float acc = b1[j];
    for (int k = 0; k < HD; ++k)
      acc += mem[m * HD + k] * w[k] + a1[m * HD + k] * w[HD + k] + a2[m * HD + k] * w[2 * HD + k];
    hid[m][j] = tanhf(acc);
  }
  __syncthreads();
  if (tid < BATCH) {
    float l0 = b2[0], l1 = b2[1];
    for (int k = 0; k < HD; ++k) {
      l0 += hid[tid][k] * W2[k];
      l1 += hid[tid][k] * W2[HD + k];
    }
    float mx = fmaxf(l0, l1);
    float e0 = expf(l0 - mx), e1 = expf(l1 - mx);
    float s = e0 + e1;
    out[tid * 2 + 0] = e0 / s;
    out[tid * 2 + 1] = e1 / s;
  }
}

// ---------------------------------------------------------------------------
extern "C" void kernel_launch(void* const* d_in, const int* in_sizes, int n_in,
                              void* d_out, int out_size, void* d_ws, size_t ws_size,
                              hipStream_t stream) {
  const int*   c_p    = (const int*)d_in[0];
  const int*   c_mask = (const int*)d_in[1];
  const int*   q_p    = (const int*)d_in[2];
  const int*   q_mask = (const int*)d_in[3];
  const int*   a1_p   = (const int*)d_in[4];
  const int*   a1_m   = (const int*)d_in[5];
  const int*   a2_p   = (const int*)d_in[6];
  const int*   a2_m   = (const int*)d_in[7];
  const float* embed  = (const float*)d_in[8];
  const float* ctx_Wih = (const float*)d_in[9],  *ctx_Whh = (const float*)d_in[10];
  const float* ctx_bih = (const float*)d_in[11], *ctx_bhh = (const float*)d_in[12];
  const float* q_Wih   = (const float*)d_in[13], *q_Whh   = (const float*)d_in[14];
  const float* q_bih   = (const float*)d_in[15], *q_bhh   = (const float*)d_in[16];
  const float* ans_Wih = (const float*)d_in[17], *ans_Whh = (const float*)d_in[18];
  const float* ans_bih = (const float*)d_in[19], *ans_bhh = (const float*)d_in[20];
  const float* att_Wih = (const float*)d_in[21], *att_Whh = (const float*)d_in[22];
  const float* att_bih = (const float*)d_in[23], *att_bhh = (const float*)d_in[24];
  const float* mem_Wih = (const float*)d_in[25], *mem_Whh = (const float*)d_in[26];
  const float* mem_bih = (const float*)d_in[27], *mem_bhh = (const float*)d_in[28];
  const float* gate_W1 = (const float*)d_in[29], *gate_b1 = (const float*)d_in[30];
  const float* gate_W2 = (const float*)d_in[31], *gate_b2 = (const float*)d_in[32];
  const float* out_W1  = (const float*)d_in[33], *out_b1  = (const float*)d_in[34];
  const float* out_W2  = (const float*)d_in[35], *out_b2  = (const float*)d_in[36];

  // workspace carve
  char* p = (char*)d_ws;
  auto take = [&](size_t bytes) -> char* {
    char* r = p;
    p += (bytes + 255) & ~(size_t)255;
    return r;
  };
  const size_t WN = (size_t)G3 * HD;  // 196608 per GRU matrix
  __bf16* bCtxI = (__bf16*)take(WN * 2);
  __bf16* bCtxH = (__bf16*)take(WN * 2);
  __bf16* bQI   = (__bf16*)take(WN * 2);
  __bf16* bQH   = (__bf16*)take(WN * 2);
  __bf16* bAnI  = (__bf16*)take(WN * 2);
  __bf16* bAnH  = (__bf16*)take(WN * 2);
  __bf16* bAtI  = (__bf16*)take(WN * 2);
  __bf16* bAtH  = (__bf16*)take(WN * 2);
  __bf16* bMeI  = (__bf16*)take(WN * 2);
  __bf16* bMeH  = (__bf16*)take(WN * 2);
  __bf16* bGW1  = (__bf16*)take((size_t)HD * G7 * 2);
  float* enc_ctx = (float*)take((size_t)BATCH * NSENT * HD * 4);  // [2560, 256]
  float* enc_q   = (float*)take((size_t)BATCH * HD * 4);
  float* enc_a1  = (float*)take((size_t)BATCH * HD * 4);
  float* enc_a2  = (float*)take((size_t)BATCH * HD * 4);
  float* attn_gi = (float*)take((size_t)BATCH * NSENT * G3 * 4);  // [2560, 768]
  float* mem_out = (float*)take((size_t)BATCH * HD * 4);

  // weight conversions (bf16)
  const int WNi = (int)WN, GWn = HD * G7;
  int nb = (WNi + 255) / 256;
  to_bf16_k<<<nb, 256, 0, stream>>>(ctx_Wih, bCtxI, WNi);
  to_bf16_k<<<nb, 256, 0, stream>>>(ctx_Whh, bCtxH, WNi);
  to_bf16_k<<<nb, 256, 0, stream>>>(q_Wih, bQI, WNi);
  to_bf16_k<<<nb, 256, 0, stream>>>(q_Whh, bQH, WNi);
  to_bf16_k<<<nb, 256, 0, stream>>>(ans_Wih, bAnI, WNi);
  to_bf16_k<<<nb, 256, 0, stream>>>(ans_Whh, bAnH, WNi);
  to_bf16_k<<<nb, 256, 0, stream>>>(att_Wih, bAtI, WNi);
  to_bf16_k<<<nb, 256, 0, stream>>>(att_Whh, bAtH, WNi);
  to_bf16_k<<<nb, 256, 0, stream>>>(mem_Wih, bMeI, WNi);
  to_bf16_k<<<nb, 256, 0, stream>>>(mem_Whh, bMeH, WNi);
  to_bf16_k<<<(GWn + 255) / 256, 256, 0, stream>>>(gate_W1, bGW1, GWn);

  // fused GRU scans
  gru_scan_k<<<(BATCH * NSENT) / SROWS, 512, 0, stream>>>(
      c_p, c_mask, embed, bCtxI, bCtxH, ctx_bih, ctx_bhh, enc_ctx, BATCH * NSENT, SLEN);
  gru_scan_k<<<BATCH / SROWS, 512, 0, stream>>>(
      q_p, q_mask, embed, bQI, bQH, q_bih, q_bhh, enc_q, BATCH, QLEN);
  gru_scan_k<<<BATCH / SROWS, 512, 0, stream>>>(
      a1_p, a1_m, embed, bAnI, bAnH, ans_bih, ans_bhh, enc_a1, BATCH, ALEN);
  gru_scan_k<<<BATCH / SROWS, 512, 0, stream>>>(
      a2_p, a2_m, embed, bAnI, bAnH, ans_bih, ans_bhh, enc_a2, BATCH, ALEN);

  // attention input projection: [2560,768] = enc_ctx @ attn_Wih^T + attn_bih
  gemm_bias_k<<<dim3((BATCH * NSENT) / 16, G3 / 64), 128, 0, stream>>>(
      enc_ctx, bAtI, att_bih, attn_gi, BATCH * NSENT, G3, HD);

  // episodic memory (persistent single workgroup)
  episodic_k<<<1, 512, 0, stream>>>(enc_ctx, enc_q, attn_gi, bGW1, gate_b1,
                                    gate_W2, gate_b2, bAtH, att_bhh, bMeI, bMeH,
                                    mem_bih, mem_bhh, mem_out);

  // output head + softmax
  final_head_k<<<1, 512, 0, stream>>>(mem_out, enc_a1, enc_a2, out_W1, out_b1,
                                      out_W2, out_b2, (float*)d_out);
}
